// TemporalAttention_64570538328077
// MI455X (gfx1250) — matrix operand order repaired and checked by reading, not compiled, not run on hardware
//
#include <hip/hip_runtime.h>

#define UNITS 128
#define QDIM 256
#define VDIM 256
#define BATCH 128
#define SEQ 4096
#define NSPLIT 8
#define ROWS_PER_SPLIT (SEQ / NSPLIT)     // 512
#define TILE_M 16
#define TILES (ROWS_PER_SPLIT / TILE_M)   // 32
#define LDS_STRIDE (VDIM + 4)             // 260 floats: 16B-aligned rows, 4-bank skew

typedef __attribute__((ext_vector_type(16))) _Float16 v16h;
typedef __attribute__((ext_vector_type(8)))  float    v8f;

// ---- CDNA5 async global->LDS copy (ASYNCcnt) with portable fallback --------
#if defined(__has_builtin)
#  if __has_builtin(__builtin_amdgcn_global_load_async_to_lds_b128) && \
      __has_builtin(__builtin_amdgcn_s_wait_asynccnt)
#    define USE_ASYNC_COPY 1
#  endif
#endif
#ifndef USE_ASYNC_COPY
#  define USE_ASYNC_COPY 0
#endif

#if USE_ASYNC_COPY
// Builtin signature (from clang diagnostic): first param is
//   "__attribute__((__vector_size__(4*sizeof(int)))) int __device__ *"
// i.e. pointer to 16-byte int vector in addrspace(1); LDS side is addrspace(3).
typedef int v4i_vs __attribute__((vector_size(4 * sizeof(int))));
typedef __attribute__((address_space(1))) v4i_vs gv4i;
typedef __attribute__((address_space(3))) v4i_vs lv4i;

__device__ __forceinline__ void async_copy_b128(const void* gsrc, void* ldst) {
  // LDS generic pointers carry the LDS byte offset in their low 32 bits.
  __builtin_amdgcn_global_load_async_to_lds_b128(
      (gv4i*)(unsigned long long)gsrc,
      (lv4i*)(unsigned int)(unsigned long long)ldst,
      0, 0);
}
#endif

__device__ __forceinline__ void stage_tile(const float* __restrict__ src,
                                           float (*dst)[LDS_STRIDE], int tid) {
#if USE_ASYNC_COPY
#pragma unroll
  for (int j = 0; j < 4; ++j) {
    const int f4  = tid + 256 * j;        // 1024 float4 chunks per 16x256 tile
    const int row = f4 >> 6;
    const int c4  = f4 & 63;
    async_copy_b128(src + (size_t)f4 * 4, &dst[row][c4 * 4]);
  }
#else
  const float4* s4 = (const float4*)src;
#pragma unroll
  for (int j = 0; j < 4; ++j) {
    const int f4  = tid + 256 * j;
    const int row = f4 >> 6;
    const int c4  = f4 & 63;
    *(float4*)(&dst[row][c4 * 4]) = s4[f4];
  }
#endif
}

__device__ __forceinline__ void wait_stage() {
#if USE_ASYNC_COPY
  __builtin_amdgcn_s_wait_asynccnt(0);    // this wave's async DMAs landed in LDS
#endif
}

// ---------------------------------------------------------------------------
// Phase 0: q_proj[b,u] = query[b,:] . W1[:,u] + b1[u]    (tiny GEMV, 8 MFLOP)
// ---------------------------------------------------------------------------
__global__ __launch_bounds__(128)
void qproj_kernel(const float* __restrict__ query, const float* __restrict__ W1,
                  const float* __restrict__ b1, float* __restrict__ qproj) {
  const int b = blockIdx.x;
  const int u = threadIdx.x;
  const float* q = query + (size_t)b * QDIM;
  float acc = b1[u];
#pragma unroll 8
  for (int k = 0; k < QDIM; ++k)
    acc += q[k] * W1[(size_t)k * UNITS + u];
  qproj[(size_t)b * UNITS + u] = acc;
}

// ---------------------------------------------------------------------------
// Phase 1: fused   score = tanh(qp + V@W2 + b2) . Wv + bv
//          online softmax + context accumulation, single HBM pass over values,
//          double-buffered async LDS staging.
// grid = (NSPLIT, BATCH), 256 threads = 8 waves.
// Wave w owns units [16w, 16w+16); its W2 slice lives in registers (B operand).
// ---------------------------------------------------------------------------
__global__ __launch_bounds__(256)
void attn_main_kernel(const float* __restrict__ values,
                      const float* __restrict__ W2,
                      const float* __restrict__ b2,
                      const float* __restrict__ Wv,
                      const float* __restrict__ bvp,
                      const float* __restrict__ qproj,
                      float* __restrict__ scores,
                      float* __restrict__ pm,
                      float* __restrict__ pl,
                      float* __restrict__ pc) {
  __shared__ float As[2][TILE_M][LDS_STRIDE];   // double-buffered 16x256 fp32 tiles
  __shared__ float score_s[TILE_M];

  const int split = blockIdx.x;
  const int b     = blockIdx.y;
  const int tid   = threadIdx.x;
  const int wave  = tid >> 5;
  const int lane  = tid & 31;
  const int hf    = lane >> 4;      // 0: lanes 0-15, 1: lanes 16-31
  const int l15   = lane & 15;

  // Per-lane unit column inside this wave's 16-wide N tile.
  const int n_local = wave * 16 + l15;
  const float qpb = qproj[(size_t)b * UNITS + n_local] + b2[n_local];
  const float wv  = Wv[n_local];
  const float bv  = bvp[0];

  // --- Preload B operand (W2 columns for this wave) as f16, contiguous k order (same as A):
  //     VGPR pair p holds K = 2p,2p+1 for lanes 0-15 and K = 16+2p,17+2p for
  //     lanes 16-31. Loop-invariant: loaded once, reused for all 512 rows. ---
  v16h Breg[8];
  {
    const int koff = hf * 16;
#pragma unroll
    for (int kk = 0; kk < 8; ++kk) {
#pragma unroll
      for (int h = 0; h < 16; ++h) {
        const int p = h >> 1, bit = h & 1;
        const int k = kk * 32 + koff + 2 * p + bit;
        Breg[kk][h] = (_Float16)W2[(size_t)k * UNITS + n_local];
      }
    }
  }

  const float* vptr = values + (size_t)b * SEQ * VDIM
                             + (size_t)split * ROWS_PER_SPLIT * VDIM;

  float m_run = -3.402823466e38f;   // running max
  float l_run = 0.0f;               // running exp-sum
  float c_acc = 0.0f;               // context accumulator; this thread owns column `col`
  const int col = tid;              // VDIM == 256 == blockDim.x
  const int m_row  = l15;           // A-operand row for WMMA loads
  const int a_koff = hf * 16;       // A layout: hi lanes hold K+16..31

  int cur = 0;
  stage_tile(vptr, As[0], tid);     // prologue: tile 0 in flight

  for (int t = 0; t < TILES; ++t) {
    wait_stage();
    __syncthreads();                // As[cur] visible; previous compute finished

    // Overlap: DMA tile t+1 into the other buffer while we compute on tile t.
    if (t + 1 < TILES)
      stage_tile(vptr + (size_t)(t + 1) * TILE_M * VDIM, As[cur ^ 1], tid);
    if (tid < TILE_M) score_s[tid] = 0.0f;
    __syncthreads();

    // ---- WMMA: acc(16x16) = A(16x256,f16) x Breg(256x16,f16), fp32 acc ----
    v8f acc = {};
#pragma unroll
    for (int kk = 0; kk < 8; ++kk) {
      v16h a;
      const float* arow = &As[cur][m_row][kk * 32 + a_koff];
#pragma unroll
      for (int h = 0; h < 16; ++h) {
        const int p = h >> 1, bit = h & 1;
        const int k = 2 * p + bit;
        a[h] = (_Float16)arow[k];
      }
      acc = __builtin_amdgcn_wmma_f32_16x16x32_f16(
          /*neg_a=*/false, a, /*neg_b=*/false, Breg[kk],
          /*c_mod=*/(short)0, acc, /*reuse_a=*/false, /*reuse_b=*/false);
    }

    // ---- per-row score partials: tanh(qp + acc) * Wv, reduce over the 16
    //      lanes of each half (wave32 shfl_xor), then across waves in LDS ----
#pragma unroll
    for (int v = 0; v < 8; ++v) {
      float part = tanhf(qpb + acc[v]) * wv;
#pragma unroll
      for (int mask = 1; mask < 16; mask <<= 1)
        part += __shfl_xor(part, mask, 32);
      if (l15 == 0) atomicAdd(&score_s[v + hf * 8], part);   // ds_add_f32
    }
    __syncthreads();

    // ---- raw scores out + online softmax + context accumulation ----
    if (tid < TILE_M)
      scores[(size_t)b * SEQ + (size_t)split * ROWS_PER_SPLIT + t * TILE_M + tid] =
          score_s[tid] + bv;

    float sc[TILE_M];
    float tmax = -3.402823466e38f;
#pragma unroll
    for (int i = 0; i < TILE_M; ++i) {
      sc[i] = score_s[i] + bv;
      tmax = fmaxf(tmax, sc[i]);
    }
    const float new_m = fmaxf(m_run, tmax);
    const float scale = __expf(m_run - new_m);
    c_acc *= scale;
    l_run *= scale;
#pragma unroll
    for (int i = 0; i < TILE_M; ++i) {
      const float e = __expf(sc[i] - new_m);
      l_run += e;
      c_acc += e * As[cur][i][col];   // values tile re-read from LDS, not HBM
    }
    m_run = new_m;
    cur ^= 1;
  }

  // ---- per-(batch,split) partials ----
  const int pidx = b * NSPLIT + split;
  pc[(size_t)pidx * VDIM + col] = c_acc;
  if (tid == 0) { pm[pidx] = m_run; pl[pidx] = l_run; }
}

// ---------------------------------------------------------------------------
// Phase 2: merge the NSPLIT partials per batch -> context + final (m, 1/l)
// ---------------------------------------------------------------------------
__global__ __launch_bounds__(256)
void combine_kernel(const float* __restrict__ pm, const float* __restrict__ pl,
                    const float* __restrict__ pc, float* __restrict__ ctx_out,
                    float* __restrict__ fm, float* __restrict__ finv) {
  const int b = blockIdx.x;
  const int col = threadIdx.x;
  float m = -3.402823466e38f;
#pragma unroll
  for (int i = 0; i < NSPLIT; ++i) m = fmaxf(m, pm[b * NSPLIT + i]);
  float l = 0.0f, c = 0.0f;
#pragma unroll
  for (int i = 0; i < NSPLIT; ++i) {
    const float e = __expf(pm[b * NSPLIT + i] - m);
    l += pl[b * NSPLIT + i] * e;
    c += pc[(size_t)(b * NSPLIT + i) * VDIM + col] * e;
  }
  const float inv = 1.0f / l;
  ctx_out[(size_t)b * VDIM + col] = c * inv;
  if (col == 0) { fm[b] = m; finv[b] = inv; }
}

// ---------------------------------------------------------------------------
// Phase 3: attention_weights[b,s] = exp(score - m_b) * inv_b
// ---------------------------------------------------------------------------
__global__ __launch_bounds__(256)
void weights_kernel(const float* __restrict__ scores, const float* __restrict__ fm,
                    const float* __restrict__ finv, float* __restrict__ wout) {
  const size_t idx = (size_t)blockIdx.x * 256 + threadIdx.x;
  const int b = (int)(idx >> 12);   // / SEQ
  wout[idx] = __expf(scores[idx] - fm[b]) * finv[b];
}

// ---------------------------------------------------------------------------
extern "C" void kernel_launch(void* const* d_in, const int* in_sizes, int n_in,
                              void* d_out, int out_size, void* d_ws, size_t ws_size,
                              hipStream_t stream) {
  const float* query  = (const float*)d_in[0];
  const float* values = (const float*)d_in[1];
  const float* W1     = (const float*)d_in[2];
  const float* b1     = (const float*)d_in[3];
  const float* W2     = (const float*)d_in[4];
  const float* b2     = (const float*)d_in[5];
  const float* Wv     = (const float*)d_in[6];
  const float* bv     = (const float*)d_in[7];
  float* out = (float*)d_out;   // [context: 128*256][weights: 128*4096]
  float* ws  = (float*)d_ws;

  float* qproj  = ws;                                  // 128*128
  float* scores = qproj + BATCH * UNITS;               // 128*4096
  float* pm     = scores + (size_t)BATCH * SEQ;        // 128*8
  float* pl     = pm + BATCH * NSPLIT;                 // 128*8
  float* pc     = pl + BATCH * NSPLIT;                 // 128*8*256
  float* fm     = pc + (size_t)BATCH * NSPLIT * VDIM;  // 128
  float* finv   = fm + BATCH;                          // 128

  qproj_kernel<<<BATCH, 128, 0, stream>>>(query, W1, b1, qproj);
  attn_main_kernel<<<dim3(NSPLIT, BATCH), 256, 0, stream>>>(
      values, W2, b2, Wv, bv, qproj, scores, pm, pl, pc);
  combine_kernel<<<BATCH, VDIM, 0, stream>>>(pm, pl, pc, out, fm, finv);
  weights_kernel<<<(BATCH * SEQ) / 256, 256, 0, stream>>>(
      scores, fm, finv, out + BATCH * VDIM);
}
